// MHSAModule_2585570312341
// MI455X (gfx1250) — matrix operand order repaired and checked by reading, not compiled, not run on hardware
//
#include <hip/hip_runtime.h>
#include <hip/hip_bf16.h>

// ---------------------------------------------------------------------------
// Types for WMMA fragments (gfx1250, wave32)
// ---------------------------------------------------------------------------
typedef __bf16 bf16_t;
typedef bf16_t v16bf __attribute__((ext_vector_type(16)));
typedef float  v8f   __attribute__((ext_vector_type(8)));
typedef unsigned int u32x4 __attribute__((ext_vector_type(4)));
typedef unsigned int v4u __attribute__((ext_vector_type(4)));
typedef int v4i __attribute__((ext_vector_type(4)));
typedef int v8i __attribute__((ext_vector_type(8)));

union Frag16 {
    v16bf v;
    u32x4 q[2];
    unsigned short u[16];
};

static __device__ __forceinline__ unsigned short f2bf(float f) {
    unsigned int u = __float_as_uint(f);
    unsigned int r = u + 0x7FFFu + ((u >> 16) & 1u);   // round-to-nearest-even
    return (unsigned short)(r >> 16);
}

static __device__ __forceinline__ v8f wmma_bf16(const Frag16& a, const Frag16& b, v8f c) {
    // D = A(16x32 bf16) * B(32x16 bf16) + C(16x16 f32)
    return __builtin_amdgcn_wmma_f32_16x16x32_bf16(
        /*neg_a=*/false, a.v, /*neg_b=*/false, b.v,
        /*c_mod=*/(short)0, c, /*reuse_a=*/false, /*reuse_b=*/false);
}

// ---------------------------------------------------------------------------
// Tensor Data Mover: 2D bf16 tile load Global -> LDS
// D# built per CDNA5 ISA; tensor described as the
// tile itself (no OOB). Tracked by TENSORcnt.
// ---------------------------------------------------------------------------
static __device__ __forceinline__ void tdm_load_2d_bf16(
    unsigned lds_byte, const void* gaddr,
    unsigned tile_d0 /*elems, contiguous*/, unsigned tile_d1 /*rows*/,
    unsigned long long stride0 /*elems*/)
{
    unsigned long long ga = (unsigned long long)(size_t)gaddr;
    v4u g0;
    g0[0] = 1u;                                         // count=1 (valid), user mode
    g0[1] = lds_byte;                                   // lds_addr (bytes)
    g0[2] = (unsigned)(ga & 0xFFFFFFFFu);               // global_addr[31:0]
    g0[3] = (unsigned)((ga >> 32) & 0x01FFFFFFu)        // global_addr[56:32]
          | (2u << 30);                                 // type=2 ("image")
    v8i g1;
    g1[0] = (int)(1u << 16);                            // wg_mask=0, data_size=1 (2B)
    g1[1] = (int)((tile_d0 & 0xFFFFu) << 16);           // tensor_dim0[15:0]
    g1[2] = (int)((tile_d0 >> 16) | ((tile_d1 & 0xFFFFu) << 16));   // tdim0 hi | tdim1 lo
    g1[3] = (int)((tile_d1 >> 16) | ((tile_d0 & 0xFFFFu) << 16));   // tdim1 hi | tile_dim0
    g1[4] = (int)(tile_d1 & 0xFFFFu);                   // tile_dim1 | tile_dim2=0
    g1[5] = (int)(unsigned)(stride0 & 0xFFFFFFFFu);     // tensor_dim0_stride[31:0]
    g1[6] = (int)(unsigned)((stride0 >> 32) & 0xFFFFu); // stride0 hi | stride1 lo = 0
    g1[7] = 0;
    v4i z4 = {0, 0, 0, 0};
#if __clang_major__ >= 23
    v8i z8 = {0, 0, 0, 0, 0, 0, 0, 0};
    __builtin_amdgcn_tensor_load_to_lds(g0, g1, z4, z4, z8, 0);
#else
    __builtin_amdgcn_tensor_load_to_lds(g0, g1, z4, z4, 0);
#endif
}

static __device__ __forceinline__ unsigned lds_offset_of(const void* p) {
    // generic shared address: low 32 bits == LDS byte offset
    return (unsigned)(size_t)p;
}

#define D_MODEL 512
#define N_HEADS 8
#define D_K     64
#define SEQ_T   2048
#define BATCH   4
#define BT      (BATCH * SEQ_T)   // 8192

// ---------------------------------------------------------------------------
// Kernel 1: convert+transpose weights fp32[K][N] -> bf16 wT[N][K]
// ---------------------------------------------------------------------------
__global__ void convw_kernel(const float* __restrict__ wq, const float* __restrict__ wk,
                             const float* __restrict__ wv, const float* __restrict__ wfc,
                             unsigned short* __restrict__ wTall, unsigned short* __restrict__ wfcT)
{
    int idx = blockIdx.x * 256 + threadIdx.x;     // over 512*512
    int w   = blockIdx.y;                         // 0..3
    int k = idx >> 9, n = idx & 511;
    const float* src = (w == 0) ? wq : (w == 1) ? wk : (w == 2) ? wv : wfc;
    unsigned short b = f2bf(src[k * D_MODEL + n]);
    if (w < 3) wTall[(size_t)w * D_MODEL * D_MODEL + (size_t)n * D_MODEL + k] = b;
    else       wfcT[(size_t)n * D_MODEL + k] = b;
}

// ---------------------------------------------------------------------------
// Kernel 2: h = LayerNorm(x, eps=1e-5)*g+b + PE(t) ; write fp32 and bf16
// ---------------------------------------------------------------------------
__global__ void prep_kernel(const float* __restrict__ x,
                            const float* __restrict__ g1, const float* __restrict__ b1,
                            float* __restrict__ hf, unsigned short* __restrict__ hb)
{
    __shared__ float red[256];
    const int row = blockIdx.x;
    const int tid = threadIdx.x;
    const float* xr = x + (size_t)row * D_MODEL;
    float v0 = xr[tid], v1 = xr[tid + 256];

    red[tid] = v0 + v1; __syncthreads();
    for (int s = 128; s > 0; s >>= 1) { if (tid < s) red[tid] += red[tid + s]; __syncthreads(); }
    float mu = red[0] * (1.0f / D_MODEL);
    __syncthreads();
    float d0 = v0 - mu, d1 = v1 - mu;
    red[tid] = d0 * d0 + d1 * d1; __syncthreads();
    for (int s = 128; s > 0; s >>= 1) { if (tid < s) red[tid] += red[tid + s]; __syncthreads(); }
    float rstd = rsqrtf(red[0] * (1.0f / D_MODEL) + 1e-5f);

    const int t = row & (SEQ_T - 1);
    #pragma unroll
    for (int p = 0; p < 2; ++p) {
        int   c  = tid + p * 256;
        float dv = (p == 0 ? d0 : d1);
        float hn = dv * rstd * g1[c] + b1[c];
        int   i2 = c & ~1;
        float dt = __expf(-logf(10000.0f) * (float)i2 * (1.0f / D_MODEL));
        float pe = (c & 1) ? cosf((float)t * dt) : sinf((float)t * dt);
        float h  = hn + pe;
        hf[(size_t)row * D_MODEL + c] = h;
        hb[(size_t)row * D_MODEL + c] = f2bf(h);
    }
}

// ---------------------------------------------------------------------------
// Kernel 3: QKV GEMM.  [BT,512] x [512,512]x3 -> Q,K [B,H,T,64], Vt [B,H,64,T]
// grid (BT/16, 96/4), block 128 (4 waves); 1 wave = one 16x16 output tile.
// A-slab (16x512 bf16, 16KB) staged into LDS once per block via TDM.
// ---------------------------------------------------------------------------
__global__ void gemm_qkv_kernel(const unsigned short* __restrict__ hb,
                                const unsigned short* __restrict__ wTall,
                                unsigned short* __restrict__ Q,
                                unsigned short* __restrict__ K,
                                unsigned short* __restrict__ Vt)
{
    __shared__ unsigned short tileA[16][D_MODEL];   // 16 KB
    const int lane  = threadIdx.x & 31;
    const int wave  = threadIdx.x >> 5;
    const int mtile = blockIdx.x;                 // 0..511
    const int ntile = blockIdx.y * 4 + wave;      // 0..95  (N = 1536 = 3*512)
    const int ln = lane & 15, g = lane >> 4;
    const int mrow = mtile * 16;

    if (wave == 0) {
        tdm_load_2d_bf16(lds_offset_of(&tileA[0][0]),
                         hb + (size_t)mrow * D_MODEL,
                         D_MODEL, 16, D_MODEL);
        __builtin_amdgcn_s_wait_tensorcnt(0);
    }
    __syncthreads();

    v8f acc = {0.f,0.f,0.f,0.f,0.f,0.f,0.f,0.f};
    const unsigned short* arow = &tileA[ln][0];
    const unsigned short* brow = wTall + (size_t)(ntile * 16 + ln) * D_MODEL;

    #pragma unroll 4
    for (int kk = 0; kk < D_MODEL; kk += 32) {
        Frag16 a, b;
        a.q[0] = *(const u32x4*)(arow + kk + 8 * g);        // K = kk + 8g + e
        a.q[1] = *(const u32x4*)(arow + kk + 16 + 8 * g);   // K = kk + 16 + 8g + e
        b.q[0] = *(const u32x4*)(brow + kk + 8 * g);        // K = kk + 8g + e
        b.q[1] = *(const u32x4*)(brow + kk + 16 + 8 * g);
        acc = wmma_bf16(a, b, acc);
    }

    const int col   = ntile * 16 + ln;      // 0..1535
    const int which = col >> 9;             // 0=Q 1=K 2=V
    const int c512  = col & 511;
    const int head  = c512 >> 6;
    const int dk    = c512 & 63;
    #pragma unroll
    for (int r = 0; r < 8; ++r) {
        int rowg = mrow + r + 8 * g;
        int bb = rowg >> 11, t = rowg & (SEQ_T - 1);
        unsigned short val = f2bf(acc[r]);
        size_t bh = (size_t)(bb * N_HEADS + head);
        if (which == 0)      Q [ (bh * SEQ_T + t) * D_K + dk ] = val;
        else if (which == 1) K [ (bh * SEQ_T + t) * D_K + dk ] = val;
        else                 Vt[ (bh * D_K + dk) * SEQ_T + t ] = val;
    }
}

// ---------------------------------------------------------------------------
// Kernel 4: causal flash attention.
// grid (B*H, T/64), block 128; each wave owns one 16-row q tile.
// ---------------------------------------------------------------------------
__global__ void attn_kernel(const unsigned short* __restrict__ Q,
                            const unsigned short* __restrict__ K,
                            const unsigned short* __restrict__ Vt,
                            unsigned short* __restrict__ O)
{
    __shared__ unsigned short plds[4][16 * 32];
    const int lane  = threadIdx.x & 31;
    const int wave  = threadIdx.x >> 5;
    const int bh    = blockIdx.x;                    // 0..31
    const int qbase = (blockIdx.y * 4 + wave) * 16;  // 0..2032
    const int ln = lane & 15, g = lane >> 4;

    const unsigned short* Qbh = Q  + (size_t)bh * SEQ_T * D_K;
    const unsigned short* Kbh = K  + (size_t)bh * SEQ_T * D_K;
    const unsigned short* Vbh = Vt + (size_t)bh * D_K * SEQ_T;

    // Q fragments (two K=32 slabs of d_k=64), A layout
    Frag16 aq[2];
    {
        const unsigned short* qrow = Qbh + (size_t)(qbase + ln) * D_K;
        #pragma unroll
        for (int s = 0; s < 2; ++s) {
            aq[s].q[0] = *(const u32x4*)(qrow + 32 * s + 8 * g);
            aq[s].q[1] = *(const u32x4*)(qrow + 32 * s + 16 + 8 * g);
        }
    }

    v8f acc[4];
    float mi[8], li[8];
    #pragma unroll
    for (int nt = 0; nt < 4; ++nt) { v8f z = {0.f,0.f,0.f,0.f,0.f,0.f,0.f,0.f}; acc[nt] = z; }
    #pragma unroll
    for (int r = 0; r < 8; ++r) { mi[r] = -3.0e38f; li[r] = 0.0f; }

    const int nkb = (qbase + 16 + 31) >> 5;          // causal #32-key blocks
    for (int kb = 0; kb < nkb; ++kb) {
        const int kbase = kb * 32;

        // prefetch next key block's K/V rows (one row per lane) into cache
        const int knext = kbase + 32;
        if (kb + 1 < nkb) {
            __builtin_prefetch(Kbh + (size_t)(knext + lane) * D_K, 0, 0);
            __builtin_prefetch(Vbh + (size_t)lane * SEQ_T + knext, 0, 0);
            __builtin_prefetch(Vbh + (size_t)(lane + 32) * SEQ_T + knext, 0, 0);
        }

        // ---- S = Q K^T  (16 q x 32 keys, two 16-wide n chunks) ----
        v8f s[2];
        { v8f z = {0.f,0.f,0.f,0.f,0.f,0.f,0.f,0.f}; s[0] = z; s[1] = z; }
        #pragma unroll
        for (int half = 0; half < 2; ++half) {
            const unsigned short* krow = Kbh + (size_t)(kbase + half * 16 + ln) * D_K;
            #pragma unroll
            for (int kc = 0; kc < 2; ++kc) {
                Frag16 bk;
                bk.q[0] = *(const u32x4*)(krow + 32 * kc + 8 * g);
                bk.q[1] = *(const u32x4*)(krow + 32 * kc + 16 + 8 * g);
                s[half] = wmma_bf16(aq[kc], bk, s[half]);
            }
        }

        // ---- scale + causal mask + online softmax (D-frag layout) ----
        const bool needMask = (kbase + 31 > qbase);
        #pragma unroll
        for (int r = 0; r < 8; ++r) {
            const int m  = r + 8 * g;
            const int qg = qbase + m;
            float v0 = s[0][r] * 0.125f;
            float v1 = s[1][r] * 0.125f;
            if (needMask) {
                if (kbase + ln      > qg) v0 = -1.0e9f;
                if (kbase + 16 + ln > qg) v1 = -1.0e9f;
            }
            float mx = fmaxf(v0, v1);
            #pragma unroll
            for (int off = 8; off; off >>= 1) mx = fmaxf(mx, __shfl_xor(mx, off));
            float mn    = fmaxf(mi[r], mx);
            float alpha = __expf(mi[r] - mn);
            float p0    = __expf(v0 - mn);
            float p1    = __expf(v1 - mn);
            float rs    = p0 + p1;
            #pragma unroll
            for (int off = 8; off; off >>= 1) rs += __shfl_xor(rs, off);
            li[r] = li[r] * alpha + rs;
            mi[r] = mn;
            #pragma unroll
            for (int nt = 0; nt < 4; ++nt) acc[nt][r] *= alpha;
            plds[wave][m * 32 + ln]      = f2bf(p0);
            plds[wave][m * 32 + 16 + ln] = f2bf(p1);
        }

        // wave-private LDS handoff: DS ops are in-order per wave; wait + fence
        asm volatile("s_wait_dscnt 0" ::: "memory");

        // ---- re-load P as A fragment (16x32) ----
        Frag16 ap;
        {
            const unsigned short* prow = &plds[wave][ln * 32];
            ap.q[0] = *(const u32x4*)(prow + 8 * g);
            ap.q[1] = *(const u32x4*)(prow + 16 + 8 * g);
        }

        // ---- O += P V   (V transposed: rows = d, contiguous over keys) ----
        #pragma unroll
        for (int nt = 0; nt < 4; ++nt) {
            Frag16 bv;
            const unsigned short* vrow = Vbh + (size_t)(nt * 16 + ln) * SEQ_T + kbase + 8 * g;
            bv.q[0] = *(const u32x4*)(vrow);
            bv.q[1] = *(const u32x4*)(vrow + 16);
            acc[nt] = wmma_bf16(ap, bv, acc[nt]);
        }
    }

    // ---- normalize + store O as [BT, 512] bf16 (heads interleaved) ----
    const int bb = bh >> 3, hh = bh & 7;
    #pragma unroll
    for (int r = 0; r < 8; ++r) {
        const int m = r + 8 * g;
        const float inv = 1.0f / li[r];
        const size_t rowg = (size_t)(bb * SEQ_T + qbase + m);
        #pragma unroll
        for (int nt = 0; nt < 4; ++nt)
            O[rowg * D_MODEL + hh * D_K + nt * 16 + ln] = f2bf(acc[nt][r] * inv);
    }
}

// ---------------------------------------------------------------------------
// Kernel 5: FC GEMM.  O_bf16[BT,512] x wfcT -> fc fp32 [BT,512]
// grid (BT/16, 8), block 128; A-slab staged via TDM like kernel 3.
// ---------------------------------------------------------------------------
__global__ void gemm_fc_kernel(const unsigned short* __restrict__ ob,
                               const unsigned short* __restrict__ wfcT,
                               float* __restrict__ fc)
{
    __shared__ unsigned short tileA[16][D_MODEL];   // 16 KB
    const int lane  = threadIdx.x & 31;
    const int wave  = threadIdx.x >> 5;
    const int mtile = blockIdx.x;
    const int ntile = blockIdx.y * 4 + wave;      // 0..31
    const int ln = lane & 15, g = lane >> 4;
    const int mrow = mtile * 16;

    if (wave == 0) {
        tdm_load_2d_bf16(lds_offset_of(&tileA[0][0]),
                         ob + (size_t)mrow * D_MODEL,
                         D_MODEL, 16, D_MODEL);
        __builtin_amdgcn_s_wait_tensorcnt(0);
    }
    __syncthreads();

    v8f acc = {0.f,0.f,0.f,0.f,0.f,0.f,0.f,0.f};
    const unsigned short* arow = &tileA[ln][0];
    const unsigned short* brow = wfcT + (size_t)(ntile * 16 + ln) * D_MODEL;

    #pragma unroll 4
    for (int kk = 0; kk < D_MODEL; kk += 32) {
        Frag16 a, b;
        a.q[0] = *(const u32x4*)(arow + kk + 8 * g);
        a.q[1] = *(const u32x4*)(arow + kk + 16 + 8 * g);
        b.q[0] = *(const u32x4*)(brow + kk + 8 * g);
        b.q[1] = *(const u32x4*)(brow + kk + 16 + 8 * g);
        acc = wmma_bf16(a, b, acc);
    }

    const int col = ntile * 16 + ln;
    #pragma unroll
    for (int r = 0; r < 8; ++r) {
        int rowg = mrow + r + 8 * g;
        fc[(size_t)rowg * D_MODEL + col] = acc[r];
    }
}

// ---------------------------------------------------------------------------
// Kernel 6: out = LayerNorm(fc + h, eps=1e-6)*g2+b2 + x
// ---------------------------------------------------------------------------
__global__ void epilogue_kernel(const float* __restrict__ fc, const float* __restrict__ hf,
                                const float* __restrict__ x,
                                const float* __restrict__ g2, const float* __restrict__ b2,
                                float* __restrict__ out)
{
    __shared__ float red[256];
    const int row = blockIdx.x;
    const int tid = threadIdx.x;
    const size_t base = (size_t)row * D_MODEL;
    float y0 = fc[base + tid]       + hf[base + tid];
    float y1 = fc[base + tid + 256] + hf[base + tid + 256];

    red[tid] = y0 + y1; __syncthreads();
    for (int s = 128; s > 0; s >>= 1) { if (tid < s) red[tid] += red[tid + s]; __syncthreads(); }
    float mu = red[0] * (1.0f / D_MODEL);
    __syncthreads();
    float d0 = y0 - mu, d1 = y1 - mu;
    red[tid] = d0 * d0 + d1 * d1; __syncthreads();
    for (int s = 128; s > 0; s >>= 1) { if (tid < s) red[tid] += red[tid + s]; __syncthreads(); }
    float rstd = rsqrtf(red[0] * (1.0f / D_MODEL) + 1e-6f);

    out[base + tid]       = d0 * rstd * g2[tid]       + b2[tid]       + x[base + tid];
    out[base + tid + 256] = d1 * rstd * g2[tid + 256] + b2[tid + 256] + x[base + tid + 256];
}

// ---------------------------------------------------------------------------
// Host launcher
// ---------------------------------------------------------------------------
extern "C" void kernel_launch(void* const* d_in, const int* in_sizes, int n_in,
                              void* d_out, int out_size, void* d_ws, size_t ws_size,
                              hipStream_t stream)
{
    const float* x    = (const float*)d_in[0];
    const float* ln1g = (const float*)d_in[1];
    const float* ln1b = (const float*)d_in[2];
    const float* wq   = (const float*)d_in[3];
    const float* wk   = (const float*)d_in[4];
    const float* wv   = (const float*)d_in[5];
    const float* wfc  = (const float*)d_in[6];
    const float* ln2g = (const float*)d_in[7];
    const float* ln2b = (const float*)d_in[8];
    float* out = (float*)d_out;

    // workspace layout (all offsets 256B-aligned)
    char* ws = (char*)d_ws;
    size_t off = 0;
    float*          hf    = (float*)(ws + off);          off += (size_t)BT * D_MODEL * 4;  // 16 MB
    unsigned short* hb    = (unsigned short*)(ws + off); off += (size_t)BT * D_MODEL * 2;  //  8 MB
    unsigned short* wTall = (unsigned short*)(ws + off); off += (size_t)3 * D_MODEL * D_MODEL * 2;
    unsigned short* wfcT  = (unsigned short*)(ws + off); off += (size_t)D_MODEL * D_MODEL * 2;
    unsigned short* Qb    = (unsigned short*)(ws + off); off += (size_t)BT * D_MODEL * 2;
    unsigned short* Kb    = (unsigned short*)(ws + off); off += (size_t)BT * D_MODEL * 2;
    unsigned short* Vtb   = (unsigned short*)(ws + off); off += (size_t)BT * D_MODEL * 2;
    unsigned short* Ob    = (unsigned short*)(ws + off); off += (size_t)BT * D_MODEL * 2;
    float*          fcf   = (float*)(ws + off);          off += (size_t)BT * D_MODEL * 4;

    convw_kernel<<<dim3(D_MODEL * D_MODEL / 256, 4), 256, 0, stream>>>(wq, wk, wv, wfc, wTall, wfcT);
    prep_kernel<<<dim3(BT), 256, 0, stream>>>(x, ln1g, ln1b, hf, hb);
    gemm_qkv_kernel<<<dim3(BT / 16, 24), 128, 0, stream>>>(hb, wTall, Qb, Kb, Vtb);
    attn_kernel<<<dim3(BATCH * N_HEADS, SEQ_T / 64), 128, 0, stream>>>(Qb, Kb, Vtb, Ob);
    gemm_fc_kernel<<<dim3(BT / 16, 8), 128, 0, stream>>>(Ob, wfcT, fcf);
    epilogue_kernel<<<dim3(BT), 256, 0, stream>>>(fcf, hf, x, ln2g, ln2b, out);
}
